// DeformableConv2d_66099546685554
// MI455X (gfx1250) — matrix operand order repaired and checked by reading, not compiled, not run on hardware
//
#include <hip/hip_runtime.h>

// ---------------------------------------------------------------------------
// Deformable Conv2d for MI455X (gfx1250, wave32, WMMA)
// B=8, C=128, H=W=64, O=128, K=3, KK=9, STRIDE=1, PAD=1
// Pipeline:
//   1) offset conv (direct, LDS-tiled, fp32)
//   2) w_def -> bf16 convert (flat layout already matches [O][C*9])
//   3) bilinear sample + pack bf16 im2col, layout [pixel][K] (K contiguous)
//      -> B-matrix WMMA fragments load directly from global (L2-resident),
//         no LDS transpose needed in the GEMM.
//   4) GEMM M=128, K=1152, N=32768 via v_wmma_f32_16x16x32_bf16 + bias,
//      barrier-free inner loop, 8 WMMA per K-step per wave.
// ---------------------------------------------------------------------------

#define BATCH 8
#define CIN   128
#define HH    64
#define WW    64
#define OUTC  128
#define KK    9
#define NPIX  (BATCH * HH * WW)     // 32768
#define KTOT  (CIN * KK)            // 1152
#define HWPIX (HH * WW)             // 4096

typedef __attribute__((ext_vector_type(16))) __bf16 v16bf;
typedef __attribute__((ext_vector_type(8)))  __bf16 v8bf;
typedef __attribute__((ext_vector_type(8)))  float  v8f;

// ---------------------------------------------------------------------------
// Kernel 1: offset conv. One block per (batch, 8x8 spatial tile).
// Stages the 10x10x128 input patch in LDS (51.2 KB of the 320 KB WGP LDS).
// ---------------------------------------------------------------------------
__global__ __launch_bounds__(256) void dcn_offset_conv(
    const float* __restrict__ x, const float* __restrict__ w_off,
    const float* __restrict__ b_off, float* __restrict__ offs)
{
    __shared__ float patch[CIN * 100];   // [c][10*10]
    const int bid  = blockIdx.x;         // b*64 + ty*8 + tx
    const int b    = bid >> 6;
    const int tile = bid & 63;
    const int ty   = tile >> 3, tx = tile & 7;
    const int y0   = ty * 8 - 1, x0 = tx * 8 - 1;
    const float* xb = x + (size_t)b * CIN * HWPIX;

    for (int idx = threadIdx.x; idx < CIN * 100; idx += 256) {
        const int c = idx / 100, r = idx % 100;
        const int pr = r / 10, pc = r % 10;
        const int gy = y0 + pr, gx = x0 + pc;
        float v = 0.f;
        if (gy >= 0 && gy < HH && gx >= 0 && gx < WW)
            v = xb[c * HWPIX + gy * WW + gx];
        patch[idx] = v;
    }
    __syncthreads();

    for (int idx = threadIdx.x; idx < 64 * 18; idx += 256) {
        const int p  = idx & 63;
        const int oc = idx >> 6;
        const int pr = p >> 3, pc = p & 7;
        float acc = b_off[oc];
        const float* wrow = w_off + (size_t)oc * CIN * 9;
        for (int c = 0; c < CIN; ++c) {
            const float* pp = patch + c * 100 + pr * 10 + pc;
            const float* wp = wrow + c * 9;
#pragma unroll
            for (int ky = 0; ky < 3; ++ky)
#pragma unroll
                for (int kx = 0; kx < 3; ++kx)
                    acc = fmaf(pp[ky * 10 + kx], wp[ky * 3 + kx], acc);
        }
        const int h = ty * 8 + pr, w = tx * 8 + pc;
        offs[((size_t)b * 18 + oc) * HWPIX + h * WW + w] = acc;
    }
}

// ---------------------------------------------------------------------------
// Kernel 2: w_def fp32 -> bf16.  w_def flat [O][C][3][3] == [O][C*9] already.
// ---------------------------------------------------------------------------
__global__ __launch_bounds__(256) void dcn_wcvt(
    const float* __restrict__ w_def, __bf16* __restrict__ wbf)
{
    const int i = blockIdx.x * 256 + threadIdx.x;
    if (i < OUTC * KTOT) wbf[i] = (__bf16)w_def[i];
}

// ---------------------------------------------------------------------------
// Kernel 3: bilinear sample + pack bf16, TRANSPOSED im2col: smp[pix][K].
// One thread per (b, pixel): computes all 9 tap positions/weights once
// (registers), then loops over 128 channels doing 36 gathers (L2-resident x)
// and writing 9 contiguous bf16 (18B) per channel.
// K index = c*9 + k matches the GEMM / w_def flat ordering.
// ---------------------------------------------------------------------------
__global__ __launch_bounds__(256) void dcn_sample(
    const float* __restrict__ x, const float* __restrict__ offs,
    __bf16* __restrict__ smp)
{
    const int pixg = blockIdx.x * 256 + threadIdx.x;  // 0..32767
    const int b = pixg >> 12;
    const int p = pixg & (HWPIX - 1);
    const int h = p >> 6, w = p & 63;

    const float* offb = offs + (size_t)b * 18 * HWPIX + p;

    float tw[KK][4];
    int   ti[KK][4];
#pragma unroll
    for (int k = 0; k < KK; ++k) {
        const int ky = k / 3, kx = k % 3;
        const float oy = offb[(2 * k)     * HWPIX];
        const float ox = offb[(2 * k + 1) * HWPIX];
        const float py = (float)(h - 1 + ky) + oy;
        const float px = (float)(w - 1 + kx) + ox;

        const float fy = floorf(py), fx = floorf(px);
        const float wy = py - fy,    wx = px - fx;
        const int iy0 = (int)fy, ix0 = (int)fx;
        const int iy1 = iy0 + 1, ix1 = ix0 + 1;
        const bool vy0 = (iy0 >= 0) && (iy0 < HH);
        const bool vy1 = (iy1 >= 0) && (iy1 < HH);
        const bool vx0 = (ix0 >= 0) && (ix0 < WW);
        const bool vx1 = (ix1 >= 0) && (ix1 < WW);
        const int cy0 = min(max(iy0, 0), HH - 1), cy1 = min(max(iy1, 0), HH - 1);
        const int cx0 = min(max(ix0, 0), WW - 1), cx1 = min(max(ix1, 0), WW - 1);

        tw[k][0] = (1.f - wy) * (1.f - wx) * (float)(vy0 && vx0);
        tw[k][1] = (1.f - wy) * wx         * (float)(vy0 && vx1);
        tw[k][2] = wy         * (1.f - wx) * (float)(vy1 && vx0);
        tw[k][3] = wy         * wx         * (float)(vy1 && vx1);
        ti[k][0] = cy0 * WW + cx0;
        ti[k][1] = cy0 * WW + cx1;
        ti[k][2] = cy1 * WW + cx0;
        ti[k][3] = cy1 * WW + cx1;
    }

    const float* xb = x + (size_t)b * CIN * HWPIX;
    __bf16* dst = smp + (size_t)pixg * KTOT;

    for (int c = 0; c < CIN; ++c) {
        const float* xc = xb + c * HWPIX;
        __bf16* dc = dst + c * KK;
#pragma unroll
        for (int k = 0; k < KK; ++k) {
            const float v = xc[ti[k][0]] * tw[k][0] + xc[ti[k][1]] * tw[k][1] +
                            xc[ti[k][2]] * tw[k][2] + xc[ti[k][3]] * tw[k][3];
            dc[k] = (__bf16)v;
        }
    }
}

// ---------------------------------------------------------------------------
// Kernel 4: GEMM  out[o][pix] = sum_K wbf[o][K] * smp[pix][K]  (+ bias)
// Block: 256 threads = 8 waves (4 M-groups x 2 N-groups); block tile 128x128;
// wave tile 32(M) x 64(N) = 2x4 WMMA accumulators; K stepped by 32.
// All fragments load straight from global (A: 288KB, B: 75MB, both
// L2-resident on the 192MB L2). No LDS, no barriers in the K-loop.
// A frag: lanes 0-15 K{0..7,16..23}, lanes 16-31 K{8..15,24..31}.
// B frag: lane = column (pixel), lanes 0-15 K{0..7,16..23}, lanes 16-31
//         K{8..15,24..31} = two aligned 16B loads thanks to the [pix][K] layout.
// ---------------------------------------------------------------------------
__global__ __launch_bounds__(256) void dcn_gemm(
    const __bf16* __restrict__ A,     // [128][1152]
    const __bf16* __restrict__ Bm,    // [32768][1152]
    const float*  __restrict__ bias,  // [128]
    float* __restrict__ out)          // [8][128][64][64]
{
    const int t     = threadIdx.x;
    const int lane  = t & 31;
    const int wv    = t >> 5;      // 0..7
    const int wm    = wv & 3;      // M group (rows wm*32)
    const int wn    = wv >> 2;     // N group (cols wn*64)
    const int n0    = blockIdx.x * 128;
    const int lrow  = lane & 15;
    const int lhalf = lane >> 4;   // 0 or 1

    v8f acc[2][4] = {};

    const __bf16* aptr[2];
#pragma unroll
    for (int mi = 0; mi < 2; ++mi)
        aptr[mi] = A + (size_t)(wm * 32 + mi * 16 + lrow) * KTOT + lhalf * 8;

    const __bf16* bptr[4];
#pragma unroll
    for (int ni = 0; ni < 4; ++ni)
        bptr[ni] = Bm + (size_t)(n0 + wn * 64 + ni * 16 + lrow) * KTOT + lhalf * 8;

    for (int kk = 0; kk < KTOT; kk += 32) {
        v16bf af[2];
#pragma unroll
        for (int mi = 0; mi < 2; ++mi) {
            v8bf lo = *(const v8bf*)(aptr[mi] + kk);        // K base .. +7
            v8bf hi = *(const v8bf*)(aptr[mi] + kk + 16);   // K base+16 .. +23
#pragma unroll
            for (int i = 0; i < 8; ++i) {
                af[mi][i]     = lo[i];
                af[mi][8 + i] = hi[i];
            }
        }

        v16bf bf[4];
#pragma unroll
        for (int ni = 0; ni < 4; ++ni) {
            v8bf lo = *(const v8bf*)(bptr[ni] + kk);        // K base .. +7
            v8bf hi = *(const v8bf*)(bptr[ni] + kk + 16);   // K base+16 .. +23
#pragma unroll
            for (int i = 0; i < 8; ++i) {
                bf[ni][i]     = lo[i];
                bf[ni][8 + i] = hi[i];
            }
            // prefetch next K-step of this (strided) B row into the caches
            __builtin_prefetch(bptr[ni] + kk + 32, 0, 3);
        }

#pragma unroll
        for (int mi = 0; mi < 2; ++mi)
#pragma unroll
            for (int ni = 0; ni < 4; ++ni)
                acc[mi][ni] = __builtin_amdgcn_wmma_f32_16x16x32_bf16(
                    false, af[mi], false, bf[ni],
                    (short)0, acc[mi][ni], false, false);
    }

    // ---- epilogue: + bias, scatter to NCHW ----
#pragma unroll
    for (int mi = 0; mi < 2; ++mi) {
#pragma unroll
        for (int ni = 0; ni < 4; ++ni) {
            const int n   = n0 + wn * 64 + ni * 16 + lrow;
            const int b   = n >> 12;
            const int rem = n & (HWPIX - 1);
#pragma unroll
            for (int r = 0; r < 8; ++r) {
                const int m = wm * 32 + mi * 16 + r + 8 * lhalf;
                out[(size_t)b * (OUTC * HWPIX) + (size_t)m * HWPIX + rem] =
                    acc[mi][ni][r] + bias[m];
            }
        }
    }
}

// ---------------------------------------------------------------------------
// Workspace layout (bytes):
//   [0, 2359296)              offsets fp32 [8][18][64][64]
//   [2359296, 2654208)        w_def bf16   [128][1152]
//   [2654208, 78151680)       sampled bf16 [32768][1152]  (~75 MB, L2-resident)
// ---------------------------------------------------------------------------
extern "C" void kernel_launch(void* const* d_in, const int* in_sizes, int n_in,
                              void* d_out, int out_size, void* d_ws, size_t ws_size,
                              hipStream_t stream) {
    const float* x     = (const float*)d_in[0];
    const float* w_off = (const float*)d_in[1];
    const float* b_off = (const float*)d_in[2];
    const float* w_def = (const float*)d_in[3];
    const float* b_def = (const float*)d_in[4];
    float* out = (float*)d_out;

    char* ws = (char*)d_ws;
    float*  offs = (float*)ws;
    __bf16* wbf  = (__bf16*)(ws + 2359296);
    __bf16* smp  = (__bf16*)(ws + 2654208);

    dcn_offset_conv<<<BATCH * 64, 256, 0, stream>>>(x, w_off, b_off, offs);
    dcn_wcvt<<<(OUTC * KTOT + 255) / 256, 256, 0, stream>>>(w_def, wbf);
    dcn_sample<<<NPIX / 256, 256, 0, stream>>>(x, offs, smp);
    dcn_gemm<<<NPIX / 128, 256, 0, stream>>>(wbf, smp, b_def, out);
}
